// InBlock_52587579572244
// MI455X (gfx1250) — matrix operand order repaired and checked by reading, not compiled, not run on hardware
//
#include <hip/hip_runtime.h>
#include <hip/hip_bf16.h>

// ---------------------------------------------------------------------------
// VMamba triple-block pipeline for MI455X (gfx1250, wave32, WMMA).
// GEMMs run on v_wmma_f32_16x16x32_bf16 (bf16 in, f32 accumulate), with
// b128 LDS fragment loads (A row-major, B column-major staging).
// ---------------------------------------------------------------------------

typedef __attribute__((ext_vector_type(16))) __bf16 v16bf;
typedef __attribute__((ext_vector_type(8)))  float  v8f;

constexpr int Dm    = 256;   // model dim
constexpr int DIm   = 512;   // inner dim
constexpr int NST   = 16;    // state dim
constexpr int RR    = 16;    // dt rank
constexpr int KC    = 4;     // conv taps
constexpr int BATCH = 8;
constexpr int LSEQ  = 1024;
constexpr int MROWS = BATCH * LSEQ;   // 8192 token rows
constexpr int XZW   = 2 * DIm;        // 1024
constexpr int XDW   = RR + 2 * NST;   // 48

__device__ __forceinline__ float silu_f(float x)  { return x / (1.f + __expf(-x)); }
__device__ __forceinline__ float splus_f(float x) { return (x > 20.f) ? x : log1pf(__expf(x)); }

// --------------------------- f32 -> bf16 convert ---------------------------
__global__ __launch_bounds__(256) void cvt_bf16_kernel(const float* __restrict__ in,
                                                       __bf16* __restrict__ out, int n) {
  int i = blockIdx.x * 256 + threadIdx.x;
  if (i < n) out[i] = (__bf16)in[i];
}

// ------------------------------ LayerNorm ----------------------------------
__global__ __launch_bounds__(256) void ln_kernel(const float* __restrict__ x,
                                                 const float* __restrict__ w,
                                                 const float* __restrict__ b,
                                                 __bf16* __restrict__ out) {
  int row = blockIdx.x;
  int tid = threadIdx.x;
  float v = x[(size_t)row * Dm + tid];
  float s = v, s2 = v * v;
  #pragma unroll
  for (int o = 16; o > 0; o >>= 1) {
    s  += __shfl_down(s,  o, 32);
    s2 += __shfl_down(s2, o, 32);
  }
  __shared__ float r1[8], r2[8];
  int wave = tid >> 5, lane = tid & 31;
  if (lane == 0) { r1[wave] = s; r2[wave] = s2; }
  __syncthreads();
  if (tid == 0) {
    float ts = 0.f, ts2 = 0.f;
    #pragma unroll
    for (int i = 0; i < 8; ++i) { ts += r1[i]; ts2 += r2[i]; }
    float mu  = ts / (float)Dm;
    float var = ts2 / (float)Dm - mu * mu;
    r1[0] = mu;
    r2[0] = rsqrtf(var + 1e-5f);
  }
  __syncthreads();
  float mu = r1[0], rinv = r2[0];
  out[(size_t)row * Dm + tid] = (__bf16)(((v - mu) * rinv) * w[tid] + b[tid]);
}

// ------------------------- bf16 WMMA GEMM (C = A @ B) ----------------------
// A: MxK row-major bf16, B: KxN row-major bf16, C: MxN f32.
// Requirements met by all call sites: M % 64 == 0, K % 64 == 0, N % 8 == 0.
// 64x64 macro-tile, BK=64 (two unrolled K-steps of 32), 8 waves, each wave
// owns a 16x32 output slab (two f32 accumulators, 4 WMMAs per tile).
// A staged row-major (stride 72), B staged COLUMN-major (stride 72) so every
// fragment is two ds_load_b128.
__global__ __launch_bounds__(256) void gemm_bf16_wmma(const __bf16* __restrict__ A,
                                                      const __bf16* __restrict__ B,
                                                      float* __restrict__ C,
                                                      int M, int N, int K) {
  constexpr int BM = 64, BN = 64, BK = 64;
  constexpr int SA = BK + 8;   // 72 elems = 144B row stride (16B aligned)
  constexpr int SB = BK + 8;   // 72 elems per column
  __shared__ __bf16 As[BM * SA];
  __shared__ __bf16 Bs[BN * SB];

  int tid  = threadIdx.x;
  int wave = tid >> 5;
  int lane = tid & 31;
  int lr   = lane & 15;
  int hi   = lane >> 4;
  int bm   = blockIdx.y * BM;
  int bn   = blockIdx.x * BN;
  int wm   = (wave & 3) * 16;   // wave row tile in [0,64)
  int wn   = (wave >> 2) * 32;  // wave col slab in [0,64)

  v8f acc0 = {}; v8f acc1 = {};

  for (int k0 = 0; k0 < K; k0 += BK) {
    // ---- stage A 64x64 tile: 512 uint4 chunks, 2 per thread -------------
    #pragma unroll
    for (int cc = 0; cc < 2; ++cc) {
      int c   = tid + cc * 256;          // chunk id
      int r   = c >> 3;                  // row in tile
      int oct = c & 7;                   // 8-element group along K
      const uint4* src = reinterpret_cast<const uint4*>(
          A + (size_t)(bm + r) * K + k0 + oct * 8);
      *reinterpret_cast<uint4*>(&As[r * SA + oct * 8]) = *src;
    }
    // ---- stage B 64x64 tile, column-major scatter -----------------------
    #pragma unroll
    for (int cc = 0; cc < 2; ++cc) {
      int c   = tid + cc * 256;
      int k   = c >> 3;                  // k within tile
      int oct = c & 7;                   // 8 consecutive n
      union { uint4 u; __bf16 h[8]; } val;
      if (bn + oct * 8 < N) {            // N % 8 == 0 -> whole chunk in/out
        val.u = *reinterpret_cast<const uint4*>(
            B + (size_t)(k0 + k) * N + bn + oct * 8);
      } else {
        val.u = make_uint4(0, 0, 0, 0);
      }
      #pragma unroll
      for (int j = 0; j < 8; ++j)
        Bs[(oct * 8 + j) * SB + k] = val.h[j];
    }
    // hint next K tile toward the caches (global_prefetch_b8)
    if (k0 + BK < K) {
      __builtin_prefetch(&A[(size_t)(bm + (tid >> 2)) * K + k0 + BK], 0, 1);
      __builtin_prefetch(&B[(size_t)(k0 + BK + (tid >> 4)) * N + bn], 0, 1);
    }
    __syncthreads();

    // ---- 2 K-steps of 32, all-b128 fragment loads -----------------------
    #pragma unroll
    for (int s = 0; s < 2; ++s) {
      const int ks = s * 32;
      // A frag (ISA 7.12.2): lane row = wm+lr, two 8-elem runs at
      // K = ks+hi*8 and K = ks+16+hi*8.
      const __bf16* pa = &As[(wm + lr) * SA + ks + hi * 8];
      union { uint4 u[2]; v16bf v; } af;
      af.u[0] = *reinterpret_cast<const uint4*>(pa);
      af.u[1] = *reinterpret_cast<const uint4*>(pa + 16);
      // B frags: lane column = wn+lr (and +16), two 8-elem runs at
      // K = ks+hi*8 and K = ks+16+hi*8.
      const __bf16* pb0 = &Bs[(wn + lr) * SB + ks + hi * 8];
      const __bf16* pb1 = pb0 + 16 * SB;
      union { uint4 u[2]; v16bf v; } bf0, bf1;
      bf0.u[0] = *reinterpret_cast<const uint4*>(pb0);
      bf0.u[1] = *reinterpret_cast<const uint4*>(pb0 + 16);
      bf1.u[0] = *reinterpret_cast<const uint4*>(pb1);
      bf1.u[1] = *reinterpret_cast<const uint4*>(pb1 + 16);

      acc0 = __builtin_amdgcn_wmma_f32_16x16x32_bf16(false, af.v, false, bf0.v,
                                                     (short)0, acc0, false, false);
      acc1 = __builtin_amdgcn_wmma_f32_16x16x32_bf16(false, af.v, false, bf1.v,
                                                     (short)0, acc1, false, false);
    }
    __syncthreads();
  }

  // D layout: VGPR r -> row (wm + r + hi*8), col = wn + (lane&15)
  #pragma unroll
  for (int r = 0; r < 8; ++r) {
    int gm  = bm + wm + hi * 8 + r;
    int gn0 = bn + wn + lr;
    int gn1 = gn0 + 16;
    if (gn0 < N) C[(size_t)gm * N + gn0] = acc0[r];
    if (gn1 < N) C[(size_t)gm * N + gn1] = acc1[r];
  }
}

// ----------------------- depthwise causal conv + SiLU ----------------------
__global__ __launch_bounds__(256) void conv_silu_kernel(const float* __restrict__ xz,
                                                        const float* __restrict__ Wc,
                                                        const float* __restrict__ bc,
                                                        float* __restrict__ xc,
                                                        __bf16* __restrict__ xcb,
                                                        int total) {
  int idx = blockIdx.x * 256 + threadIdx.x;
  if (idx >= total) return;
  int d = idx & (DIm - 1);
  int m = idx >> 9;           // token row
  int l = m & (LSEQ - 1);     // position within sequence
  float acc = bc[d];
  #pragma unroll
  for (int i = 0; i < KC; ++i) {
    int ll = l + i - (KC - 1);
    if (ll >= 0) acc += xz[(size_t)(m + i - (KC - 1)) * XZW + d] * Wc[d * KC + i];
  }
  float s = silu_f(acc);
  xc[idx]  = s;
  xcb[idx] = (__bf16)s;
}

// ----------------------------- dt = softplus -------------------------------
__global__ __launch_bounds__(256) void dt_kernel(const float* __restrict__ xdbl,
                                                 const float* __restrict__ Wdt,
                                                 const float* __restrict__ bdt,
                                                 float* __restrict__ dtv,
                                                 int total) {
  int idx = blockIdx.x * 256 + threadIdx.x;
  if (idx >= total) return;
  int d = idx & (DIm - 1);
  int m = idx >> 9;
  const float* dr = xdbl + (size_t)m * XDW;   // first RR entries = dt_r
  float acc = bdt[d];
  #pragma unroll
  for (int r = 0; r < RR; ++r) acc += dr[r] * Wdt[r * DIm + d];
  dtv[idx] = splus_f(acc);
}

// -------------------------- selective scan (serial L) ----------------------
__global__ __launch_bounds__(128) void scan_kernel(const float* __restrict__ xc,
                                                   const float* __restrict__ dtv,
                                                   const float* __restrict__ xdbl,
                                                   const float* __restrict__ A_log,
                                                   float* __restrict__ ysc) {
  int d = blockIdx.x * 128 + threadIdx.x;
  int b = blockIdx.y;
  float A[NST], h[NST];
  #pragma unroll
  for (int n = 0; n < NST; ++n) {
    A[n] = -__expf(A_log[(size_t)d * NST + n]);
    h[n] = 0.f;
  }
  for (int l = 0; l < LSEQ; ++l) {
    size_t m = (size_t)b * LSEQ + l;
    float xt = xc[m * DIm + d];
    float dt = dtv[m * DIm + d];
    const float* Bv = xdbl + m * XDW + RR;
    const float* Cv = xdbl + m * XDW + RR + NST;
    float y = 0.f;
    #pragma unroll
    for (int n = 0; n < NST; ++n) {
      float dA = __expf(dt * A[n]);
      h[n] = dA * h[n] + (dt * Bv[n]) * xt;
      y += h[n] * Cv[n];
    }
    ysc[m * DIm + d] = y;
  }
}

// --------------------- y = (ysc + xc*Dskip) * silu(z) ----------------------
__global__ __launch_bounds__(256) void gate_kernel(const float* __restrict__ ysc,
                                                   const float* __restrict__ xc,
                                                   const float* __restrict__ Dskip,
                                                   const float* __restrict__ xz,
                                                   __bf16* __restrict__ ygb,
                                                   int total) {
  int idx = blockIdx.x * 256 + threadIdx.x;
  if (idx >= total) return;
  int d = idx & (DIm - 1);
  int m = idx >> 9;
  float z = xz[(size_t)m * XZW + DIm + d];
  float y = (ysc[idx] + xc[idx] * Dskip[d]) * silu_f(z);
  ygb[idx] = (__bf16)y;
}

// ------------------------------ combiners ----------------------------------
__global__ __launch_bounds__(256) void combine0_kernel(const float* __restrict__ z1,
                                                       const float* __restrict__ z2,
                                                       const float* __restrict__ g,
                                                       float* __restrict__ z2n,
                                                       float* __restrict__ out_z2,
                                                       int n) {
  int i = blockIdx.x * 256 + threadIdx.x;
  if (i >= n) return;
  float v = z2[i] + z1[i] + g[i];
  z2n[i] = v;
  out_z2[i] = v;
}
__global__ __launch_bounds__(256) void combine1_kernel(const float* __restrict__ z1,
                                                       const float* __restrict__ z2n,
                                                       const float* __restrict__ g,
                                                       float* __restrict__ tmp,
                                                       int n) {
  int i = blockIdx.x * 256 + threadIdx.x;
  if (i >= n) return;
  tmp[i] = z1[i] * __expf(z2n[i] + g[i]);
}
__global__ __launch_bounds__(256) void combine2_kernel(const float* __restrict__ tmp,
                                                       const float* __restrict__ z2n,
                                                       const float* __restrict__ g,
                                                       float* __restrict__ out_z1,
                                                       int n) {
  int i = blockIdx.x * 256 + threadIdx.x;
  if (i >= n) return;
  out_z1[i] = tmp[i] + z2n[i] + g[i];
}

// ---------------------------------------------------------------------------
extern "C" void kernel_launch(void* const* d_in, const int* in_sizes, int n_in,
                              void* d_out, int out_size, void* d_ws, size_t ws_size,
                              hipStream_t stream) {
  (void)in_sizes; (void)n_in; (void)out_size; (void)ws_size;

  const float* z1    = (const float*)d_in[0];
  const float* z2    = (const float*)d_in[1];
  const float* ln_w  = (const float*)d_in[2];
  const float* ln_b  = (const float*)d_in[3];
  const float* Win   = (const float*)d_in[4];
  const float* Wconv = (const float*)d_in[5];
  const float* bconv = (const float*)d_in[6];
  const float* Wx    = (const float*)d_in[7];
  const float* Wdt   = (const float*)d_in[8];
  const float* bdt   = (const float*)d_in[9];
  const float* A_log = (const float*)d_in[10];
  const float* Dskip = (const float*)d_in[11];
  const float* Wout  = (const float*)d_in[12];

  const size_t E = (size_t)MROWS * Dm;          // 2,097,152 elements
  float* out_z1 = (float*)d_out;
  float* out_z2 = out_z1 + E;

  // carve scratch out of d_ws
  char*  wsp = (char*)d_ws;
  size_t off = 0;
  auto alloc = [&](size_t bytes) -> char* {
    char* p = wsp + off;
    off += (bytes + 255) & ~(size_t)255;
    return p;
  };
  __bf16* xnb   = (__bf16*)alloc((size_t)MROWS * Dm  * sizeof(__bf16));
  __bf16* Winb  = (__bf16*)alloc((size_t)Dm   * XZW  * sizeof(__bf16));
  __bf16* Wxb   = (__bf16*)alloc((size_t)DIm  * XDW  * sizeof(__bf16));
  __bf16* Woutb = (__bf16*)alloc((size_t)DIm  * Dm   * sizeof(__bf16));
  float*  xz    = (float*) alloc((size_t)MROWS * XZW * sizeof(float));
  float*  xc    = (float*) alloc((size_t)MROWS * DIm * sizeof(float));
  __bf16* xcb   = (__bf16*)alloc((size_t)MROWS * DIm * sizeof(__bf16));
  float*  xdbl  = (float*) alloc((size_t)MROWS * XDW * sizeof(float));
  float*  dtv   = (float*) alloc((size_t)MROWS * DIm * sizeof(float));
  float*  ysc   = (float*) alloc((size_t)MROWS * DIm * sizeof(float));
  __bf16* ygb   = (__bf16*)alloc((size_t)MROWS * DIm * sizeof(__bf16));
  float*  gout  = (float*) alloc(E * sizeof(float));
  float*  z2n   = (float*) alloc(E * sizeof(float));
  float*  tmp   = (float*) alloc(E * sizeof(float));

  const int ED  = MROWS * DIm;   // 4,194,304
  const dim3 blk256(256), blk128(128);

  auto run_block = [&](int i, const float* x, float* g) {
    const float* lw   = ln_w  + (size_t)i * Dm;
    const float* lb   = ln_b  + (size_t)i * Dm;
    const float* Wi   = Win   + (size_t)i * Dm  * XZW;
    const float* Wc   = Wconv + (size_t)i * DIm * KC;
    const float* bc   = bconv + (size_t)i * DIm;
    const float* Wxi  = Wx    + (size_t)i * DIm * XDW;
    const float* Wdti = Wdt   + (size_t)i * RR  * DIm;
    const float* bdti = bdt   + (size_t)i * DIm;
    const float* Ai   = A_log + (size_t)i * DIm * NST;
    const float* Di   = Dskip + (size_t)i * DIm;
    const float* Wo   = Wout  + (size_t)i * DIm * Dm;

    cvt_bf16_kernel<<<(Dm * XZW + 255) / 256, blk256, 0, stream>>>(Wi,  Winb,  Dm * XZW);
    cvt_bf16_kernel<<<(DIm * XDW + 255) / 256, blk256, 0, stream>>>(Wxi, Wxb,  DIm * XDW);
    cvt_bf16_kernel<<<(DIm * Dm  + 255) / 256, blk256, 0, stream>>>(Wo,  Woutb, DIm * Dm);

    ln_kernel<<<MROWS, blk256, 0, stream>>>(x, lw, lb, xnb);
    gemm_bf16_wmma<<<dim3(XZW / 64, MROWS / 64), blk256, 0, stream>>>(
        xnb, Winb, xz, MROWS, XZW, Dm);
    conv_silu_kernel<<<ED / 256, blk256, 0, stream>>>(xz, Wc, bc, xc, xcb, ED);
    gemm_bf16_wmma<<<dim3((XDW + 63) / 64, MROWS / 64), blk256, 0, stream>>>(
        xcb, Wxb, xdbl, MROWS, XDW, DIm);
    dt_kernel<<<ED / 256, blk256, 0, stream>>>(xdbl, Wdti, bdti, dtv, ED);
    scan_kernel<<<dim3(DIm / 128, BATCH), blk128, 0, stream>>>(xc, dtv, xdbl, Ai, ysc);
    gate_kernel<<<ED / 256, blk256, 0, stream>>>(ysc, xc, Di, xz, ygb, ED);
    gemm_bf16_wmma<<<dim3(Dm / 64, MROWS / 64), blk256, 0, stream>>>(
        ygb, Woutb, g, MROWS, Dm, DIm);
  };

  const int gE = (int)((E + 255) / 256);

  // z2' = z2 + blk(0, z1)
  run_block(0, z1, gout);
  combine0_kernel<<<gE, blk256, 0, stream>>>(z1, z2, gout, z2n, out_z2, (int)E);

  // z1' = z1 * exp(blk(1, z2')) + blk(2, z2')
  run_block(1, z2n, gout);
  combine1_kernel<<<gE, blk256, 0, stream>>>(z1, z2n, gout, tmp, (int)E);
  run_block(2, z2n, gout);
  combine2_kernel<<<gE, blk256, 0, stream>>>(tmp, z2n, gout, out_z1, (int)E);
}
